// Attention_6605659701893
// MI455X (gfx1250) — matrix operand order repaired and checked by reading, not compiled, not run on hardware
//
#include <hip/hip_runtime.h>

typedef __attribute__((ext_vector_type(16))) _Float16 v16h;
typedef __attribute__((ext_vector_type(2)))  _Float16 v2h;
typedef __attribute__((ext_vector_type(8)))  float    v8f;
typedef __attribute__((ext_vector_type(4)))  float    v4f;

#define SEQ   2048
#define HD    64
#define QB    128   // query rows per workgroup
#define KB    64    // keys per kv iteration
#define NWAVE 8

// 1/sqrt(64) * log2(e): QK^T scores land directly in the log2 domain,
// so softmax uses raw v_exp_f32 (exp2) with no per-element multiply.
#define QSCALE 0.18033688011112042f

// DPP row-rotate (16-lane row) allreduce helpers: pure VALU, no LDS, no waits.
template <int N>
__device__ __forceinline__ float row_ror(float x) {
    int t = __builtin_amdgcn_mov_dpp(__builtin_bit_cast(int, x),
                                     0x120 | N, 0xf, 0xf, true);
    return __builtin_bit_cast(float, t);
}
__device__ __forceinline__ float row_allmax(float x) {
    x = fmaxf(x, row_ror<8>(x));
    x = fmaxf(x, row_ror<4>(x));
    x = fmaxf(x, row_ror<2>(x));
    x = fmaxf(x, row_ror<1>(x));
    return x;
}
__device__ __forceinline__ float row_allsum(float x) {
    x += row_ror<8>(x);
    x += row_ror<4>(x);
    x += row_ror<2>(x);
    x += row_ror<1>(x);
    return x;
}

// Online-softmax update for 8 rows x 64 keys (log2 domain). MASKED is a
// compile-time-foldable flag: only the diagonal KV block pays for cndmasks.
#define SOFTMAX_BODY(MASKED)                                                   \
    do {                                                                       \
        _Pragma("unroll")                                                      \
        for (int r = 0; r < 8; ++r) {                                          \
            const int qi = q0 + r + 8 * h;                                     \
            float e0 = s[0][r], e1 = s[1][r], e2 = s[2][r], e3 = s[3][r];      \
            if (MASKED) {                                                      \
                e0 = (kbase + ln16      > qi) ? -3.0e38f : e0;                 \
                e1 = (kbase + 16 + ln16 > qi) ? -3.0e38f : e1;                 \
                e2 = (kbase + 32 + ln16 > qi) ? -3.0e38f : e2;                 \
                e3 = (kbase + 48 + ln16 > qi) ? -3.0e38f : e3;                 \
            }                                                                  \
            float rm    = row_allmax(fmaxf(fmaxf(e0, e1), fmaxf(e2, e3)));     \
            float mnew  = fmaxf(mrow[r], rm);                                  \
            float alpha = __builtin_amdgcn_exp2f(mrow[r] - mnew);              \
            float p0 = __builtin_amdgcn_exp2f(e0 - mnew);                      \
            float p1 = __builtin_amdgcn_exp2f(e1 - mnew);                      \
            float p2 = __builtin_amdgcn_exp2f(e2 - mnew);                      \
            float p3 = __builtin_amdgcn_exp2f(e3 - mnew);                      \
            float rs = row_allsum((p0 + p1) + (p2 + p3));                      \
            lrow[r] = lrow[r] * alpha + rs;                                    \
            mrow[r] = mnew;                                                    \
            o[0][r] *= alpha; o[1][r] *= alpha;                                \
            o[2][r] *= alpha; o[3][r] *= alpha;                                \
            _Float16* prow = &Plds[wv][(r + 8 * h) * KB + ln16];               \
            prow[0]  = (_Float16)p0;  prow[16] = (_Float16)p1;                 \
            prow[32] = (_Float16)p2;  prow[48] = (_Float16)p3;                 \
        }                                                                      \
    } while (0)

__global__ __launch_bounds__(256)
void fa_causal_f16wmma(const float* __restrict__ Q, const float* __restrict__ K,
                       const float* __restrict__ V, float* __restrict__ Out) {
    __shared__ _Float16 Klds[KB * HD];            // [key][d]   row-major f16
    __shared__ _Float16 Vt[HD * KB];              // [d][key]   transposed f16
    __shared__ _Float16 Plds[NWAVE][16 * KB];     // per-wave P (16 x 64)

    const int tid  = threadIdx.x;
    const int lane = tid & 31;
    const int ln16 = lane & 15;
    const int h    = lane >> 4;                       // half-wave group (0/1)
    const int wv   = __builtin_amdgcn_readfirstlane(tid) >> 5;  // scalar wave id

    const int bh  = blockIdx.y;          // batch*head
    const int q0b = blockIdx.x * QB;     // workgroup query base
    const int q0  = q0b + wv * 16;       // this wave's query tile base (scalar)

    const size_t base = (size_t)bh * SEQ * HD;

    // ---- load Q fragments (A layout), scale*log2e folded in ----
    v16h qa[2];
    {
        const float* qrow = Q + base + (size_t)(q0 + ln16) * HD;
#pragma unroll
        for (int dc = 0; dc < 2; ++dc) {
            v4f f0 = *(const v4f*)(qrow + dc * 32 + 8 * h);
            v4f f1 = *(const v4f*)(qrow + dc * 32 + 8 * h + 4);
            v4f f2 = *(const v4f*)(qrow + dc * 32 + 16 + 8 * h);
            v4f f3 = *(const v4f*)(qrow + dc * 32 + 16 + 8 * h + 4);
#pragma unroll
            for (int e = 0; e < 4; ++e) {
                qa[dc][e]      = (_Float16)(f0[e] * QSCALE);
                qa[dc][4 + e]  = (_Float16)(f1[e] * QSCALE);
                qa[dc][8 + e]  = (_Float16)(f2[e] * QSCALE);
                qa[dc][12 + e] = (_Float16)(f3[e] * QSCALE);
            }
        }
    }

    v8f o[4] = {{}, {}, {}, {}};         // O accum: 4 chunks of 16 columns
    float mrow[8], lrow[8];
#pragma unroll
    for (int r = 0; r < 8; ++r) { mrow[r] = -3.0e38f; lrow[r] = 0.0f; }

    const int nkv = q0b / KB + QB / KB;  // causal bound for this workgroup
    const int kr  = tid >> 2;            // staging key row 0..63
    const int dp  = (tid & 3) * 16;      // staging d offset (16 floats each)

    for (int it = 0; it < nkv; ++it) {
        const int kbase = it * KB;

        // ---- stage K (row-major, packed b32 stores) and V (transposed) ----
        {
            const float* kg = K + base + (size_t)(kbase + kr) * HD + dp;
            const float* vg = V + base + (size_t)(kbase + kr) * HD + dp;
            v4f kf[4], vf[4];
#pragma unroll
            for (int g = 0; g < 4; ++g) {
                kf[g] = *(const v4f*)(kg + 4 * g);
                vf[g] = *(const v4f*)(vg + 4 * g);
            }
#pragma unroll
            for (int g = 0; g < 4; ++g) {
#pragma unroll
                for (int i = 0; i < 2; ++i) {   // packed f16 pair -> b32 store
                    v2h p;
                    p[0] = (_Float16)kf[g][2 * i];
                    p[1] = (_Float16)kf[g][2 * i + 1];
                    *(v2h*)&Klds[kr * HD + dp + 4 * g + 2 * i] = p;
                }
#pragma unroll
                for (int e = 0; e < 4; ++e)     // transposed: strided b16 store
                    Vt[(dp + 4 * g + e) * KB + kr] = (_Float16)vf[g][e];
            }
            if (it + 1 < nkv) {   // prefetch next KV block (keep in cache)
                __builtin_prefetch(kg + KB * HD, 0, 3);
                __builtin_prefetch(vg + KB * HD, 0, 3);
            }
        }
        __syncthreads();

        if (kbase < q0 + 16) {           // scalar causal skip (EXEC untouched)
            // ---- S = (Q K^T) for four 16-key tiles ----
            v8f s[4];
#pragma unroll
            for (int st = 0; st < 4; ++st) {
                v8f acc = {};
#pragma unroll
                for (int dc = 0; dc < 2; ++dc) {
                    v16h kb;             // B fragment: K^T, contraction = head dim
#pragma unroll
                    for (int j = 0; j < 8; ++j) {
                        int kk = dc * 32 + 16 * (j >> 2) + 8 * h + 2 * (j & 3);
                        v2h p = *(const v2h*)&Klds[(16 * st + ln16) * HD + kk];
                        kb[2 * j] = p[0]; kb[2 * j + 1] = p[1];
                    }
                    acc = __builtin_amdgcn_wmma_f32_16x16x32_f16(
                              false, qa[dc], false, kb, (short)0, acc, false, false);
                }
                s[st] = acc;
            }

            // ---- online softmax: masked path only on the diagonal block ----
            if (kbase + KB > q0) {       // scalar: wave's final (diagonal) block
                SOFTMAX_BODY(true);
            } else {                     // interior block: no causal cndmasks
                SOFTMAX_BODY(false);
            }
            asm volatile("s_wait_dscnt 0" ::: "memory");

            // ---- re-read P in A-fragment layout (two 16x32 fragments) ----
            v16h pa[2];
#pragma unroll
            for (int pk = 0; pk < 2; ++pk)
#pragma unroll
                for (int j = 0; j < 8; ++j) {
                    int kk = pk * 32 + 16 * (j >> 2) + 8 * h + 2 * (j & 3);
                    v2h p = *(const v2h*)&Plds[wv][ln16 * KB + kk];
                    pa[pk][2 * j] = p[0]; pa[pk][2 * j + 1] = p[1];
                }

            // ---- O += P V  (4 column chunks x 2 key halves) ----
#pragma unroll
            for (int c = 0; c < 4; ++c) {
#pragma unroll
                for (int pk = 0; pk < 2; ++pk) {
                    v16h vb;             // B fragment from transposed V
#pragma unroll
                    for (int j = 0; j < 8; ++j) {
                        int kk = pk * 32 + 16 * (j >> 2) + 8 * h + 2 * (j & 3);
                        v2h p = *(const v2h*)&Vt[(16 * c + ln16) * KB + kk];
                        vb[2 * j] = p[0]; vb[2 * j + 1] = p[1];
                    }
                    o[c] = __builtin_amdgcn_wmma_f32_16x16x32_f16(
                               false, pa[pk], false, vb, (short)0, o[c], false, false);
                }
            }
        }
        __syncthreads();
    }

    // ---- epilogue: normalize and store fp32 ----
#pragma unroll
    for (int r = 0; r < 8; ++r) {
        const int qi  = q0 + r + 8 * h;
        const float inv = 1.0f / lrow[r];
        float* orow = Out + base + (size_t)qi * HD + ln16;
#pragma unroll
        for (int c = 0; c < 4; ++c)
            orow[16 * c] = o[c][r] * inv;
    }
}

extern "C" void kernel_launch(void* const* d_in, const int* in_sizes, int n_in,
                              void* d_out, int out_size, void* d_ws, size_t ws_size,
                              hipStream_t stream) {
    const float* q = (const float*)d_in[0];
    const float* k = (const float*)d_in[1];
    const float* v = (const float*)d_in[2];
    float* out = (float*)d_out;
    (void)in_sizes; (void)n_in; (void)out_size; (void)d_ws; (void)ws_size;

    dim3 grid(SEQ / QB, 4 * 16);   // (q-blocks, batch*heads)
    dim3 block(256);
    fa_causal_f16wmma<<<grid, block, 0, stream>>>(q, k, v, out);
}
